// AnomalyNearestNeighbors_27410481283177
// MI455X (gfx1250) — matrix operand order repaired and checked by reading, not compiled, not run on hardware
//
#include <hip/hip_runtime.h>

// ---------------- CDNA5 (gfx1250) fused kNN: bf16 WMMA GEMM + TDM staging + top-k ----------------

typedef __attribute__((ext_vector_type(16))) __bf16 v16bf;
typedef __attribute__((ext_vector_type(8)))  float  v8f;
typedef __attribute__((ext_vector_type(4)))  unsigned int v4u;
typedef __attribute__((ext_vector_type(8)))  int  v8i;
typedef __attribute__((ext_vector_type(4)))  int  v4i;

#define MT      128           // query rows per block
#define NT      128           // fit rows per tile
#define KT      32            // K per WMMA step (bf16)
#define DDIM    512           // feature dim (F*L)
#define KSTEPS  (DDIM / KT)   // 16
#define ASTRIDE (DDIM + 8)    // 520 bf16 elems (bank-conflict pad)
#define BSTRIDE (KT + 8)      // 40 bf16 elems (TDM pad reproduces this)
#define SSTRIDE (NT + 4)      // 132 floats
#define NCHUNK  64            // N split across blockIdx.y
#define TOPK    5
#define BIGF    3.0e38f
#define BBUF_BYTES (NT * BSTRIDE * 2)   // 10240 B per B buffer

static constexpr int SMEM_BYTES =
    MT * ASTRIDE * 2 +        // A (queries, bf16)           133120 B
    2 * BBUF_BYTES +          // B double buffer (bf16)       20480 B
    MT * SSTRIDE * 4 +        // score tile (f32)             67584 B
    NT * 4;                   // x_sq tile                      512 B
                              // total                        221696 B (< 320KB WGP LDS)

#if defined(__HIP_DEVICE_COMPILE__) && __has_builtin(__builtin_amdgcn_tensor_load_to_lds)
#define HAVE_TDM 1
#else
#define HAVE_TDM 0
#endif

__device__ __forceinline__ unsigned short f2bf(float f) {
  unsigned u = __float_as_uint(f);
  u += 0x7FFFu + ((u >> 16) & 1u);        // round-to-nearest-even
  return (unsigned short)(u >> 16);
}

// A fragment 16x32 bf16 (ISA 7.12.2): lane<16 -> K 0..7 & 16..23 ; lane>=16 -> K 8..15 & 24..31
__device__ __forceinline__ v16bf load_a_frag(const unsigned short* p) {
  union { v16bf v; uint4 u[2]; } f;
  f.u[0] = *(const uint4*)(p);
  f.u[1] = *(const uint4*)(p + 16);
  return f.v;
}
// B fragment 32x16 bf16: lane<16 -> K 0..7 & 16..23 of col(lane); lane>=16 -> K 8..15 & 24..31
__device__ __forceinline__ v16bf load_b_frag(const unsigned short* p) {
  union { v16bf v; uint4 u[2]; } f;
  f.u[0] = *(const uint4*)(p);
  f.u[1] = *(const uint4*)(p + 16);
  return f.v;
}

__device__ __forceinline__ void topk_insert(float (&ts)[TOPK], int (&ti)[TOPK],
                                            float s, int id) {
  if (s >= ts[4]) return;
  if (s >= ts[3]) { ts[4] = s; ti[4] = id; return; }
  ts[4] = ts[3]; ti[4] = ti[3];
  if (s >= ts[2]) { ts[3] = s; ti[3] = id; return; }
  ts[3] = ts[2]; ti[3] = ti[2];
  if (s >= ts[1]) { ts[2] = s; ti[2] = id; return; }
  ts[2] = ts[1]; ti[2] = ti[1];
  if (s >= ts[0]) { ts[1] = s; ti[1] = id; return; }
  ts[1] = ts[0]; ti[1] = ti[0];
  ts[0] = s; ti[0] = id;
}

#if HAVE_TDM
// Issue one TDM DMA: 2D bf16 tile [128 rows x 32 cols] of xb into LDS at lds_byte_off,
// with LDS padding 4 DWORDs after every 16 DWORDs -> row stride 80B == BSTRIDE*2.
// OOB rows (>= rem_rows) read as zero per D# tensor_dim bounds.
__device__ __forceinline__ void tdm_load_b_slice(const unsigned short* xb, int n0,
                                                 int rem_rows, int k0,
                                                 unsigned lds_byte_off) {
  unsigned long long ga =
      (unsigned long long)(const void*)(xb + (size_t)n0 * DDIM + k0);
  v4u g0;
  g0.x = 1u;                                            // count=1, user descriptor
  g0.y = lds_byte_off;                                  // lds_addr
  g0.z = (unsigned)(ga & 0xFFFFFFFFull);                // global_addr[31:0]
  g0.w = (unsigned)((ga >> 32) & 0x1FFFFFFu) | (2u << 30); // global_addr[56:32] | type=2
  unsigned rem = (unsigned)rem_rows;
  v8i g1;
  g1[0] = (int)((1u << 16) |     // data_size = 1 (2 bytes)
                (1u << 20) |     // pad_enable
                (3u << 22) |     // pad_interval = 16 DWORDs (64B row)
                (3u << 25));     // pad_amount   = 4 DWORDs (16B pad)
  g1[1] = (int)(((unsigned)DDIM & 0xFFFFu) << 16);        // tensor_dim0[15:0]
  g1[2] = (int)((((unsigned)DDIM) >> 16) | ((rem & 0xFFFFu) << 16)); // dim0 hi | dim1 lo
  g1[3] = (int)((rem >> 16) | ((unsigned)KT << 16));      // dim1 hi | tile_dim0 = 32
  g1[4] = (int)(NT);                                      // tile_dim1 = 128, tile_dim2 = 0
  g1[5] = (int)(DDIM);                                    // tensor_dim0_stride[31:0] = 512
  g1[6] = 0;                                              // stride hi | dim1_stride lo
  g1[7] = 0;                                              // dim1_stride hi
  v4i gz = {0, 0, 0, 0};
#if __clang_major__ >= 23
  v8i gz8 = {0, 0, 0, 0, 0, 0, 0, 0};
  __builtin_amdgcn_tensor_load_to_lds(g0, g1, gz, gz, gz8, 0);
#else
  __builtin_amdgcn_tensor_load_to_lds(g0, g1, gz, gz, 0);
#endif
}
#endif

// ---- prep: X_fit f32 -> bf16 (workspace) + row sum-of-squares; one wave32 per row ----
__global__ __launch_bounds__(256)
void knn_prep(const float* __restrict__ xfit, unsigned short* __restrict__ xb,
              float* __restrict__ xsq, int nfit) {
  const int lane = threadIdx.x & 31;
  const int row  = blockIdx.x * 8 + (threadIdx.x >> 5);
  if (row >= nfit) return;
  const size_t base = (size_t)row << 9;   // * 512
  float ss = 0.f;
#pragma unroll
  for (int i = 0; i < 4; i++) {
    const int off = lane * 4 + i * 128;
    float4 f = *(const float4*)(xfit + base + off);
    ss += f.x * f.x + f.y * f.y + f.z * f.z + f.w * f.w;
    ushort4 h;
    h.x = f2bf(f.x); h.y = f2bf(f.y); h.z = f2bf(f.z); h.w = f2bf(f.w);
    *(ushort4*)(xb + base + off) = h;
  }
#pragma unroll
  for (int d = 16; d >= 1; d >>= 1) ss += __shfl_xor(ss, d, 32);
  if (lane == 0) xsq[row] = ss;
}

// ---- main: 128x128 score tiles via v_wmma_f32_16x16x32_bf16, TDM B staging, fused top-5 ----
__global__ __launch_bounds__(256, 1)
void knn_gemm_topk(const float* __restrict__ q,
                   const unsigned short* __restrict__ xb,
                   const float* __restrict__ xsq,
                   float* __restrict__ ps, int* __restrict__ pidx,
                   int nfit, int ntiles) {
  extern __shared__ unsigned char smem[];
  unsigned short* As = (unsigned short*)smem;
  unsigned short* Bs = As + MT * ASTRIDE;
  float* Ss   = (float*)(Bs + 2 * NT * BSTRIDE);
  float* xsqt = Ss + MT * SSTRIDE;

  const int tid   = threadIdx.x;
  const int lane  = tid & 31;
  const int wid   = tid >> 5;
  const int lo    = lane & 15;
  const int hi    = lane >> 4;
  const int mw    = (wid & 3) * 32;   // wave's M base (2 sub-tiles of 16)
  const int nw    = (wid >> 2) * 64;  // wave's N base (4 sub-tiles of 16)
  const int m0    = blockIdx.x * MT;
  const int chunk = blockIdx.y;

#if HAVE_TDM
  // LDS byte offset of the B buffers (workgroup-relative; shared flat addr low 32b)
  const unsigned bs_off = (unsigned)(size_t)(void*)Bs;
#endif

  // Stage all queries for this M-tile: f32 -> bf16 into LDS (reused for every N-tile)
  for (int e = tid * 4; e < MT * DDIM; e += 256 * 4) {
    const int row = e >> 9;
    const int col = e & (DDIM - 1);
    const float4 f = *(const float4*)(q + ((size_t)(m0 + row) << 9) + col);
    ushort4 h;
    h.x = f2bf(f.x); h.y = f2bf(f.y); h.z = f2bf(f.z); h.w = f2bf(f.w);
    *(ushort4*)(As + row * ASTRIDE + col) = h;
  }

  float ts[TOPK] = {BIGF, BIGF, BIGF, BIGF, BIGF};
  int   ti[TOPK] = {0, 0, 0, 0, 0};

  for (int jt = chunk; jt < ntiles; jt += NCHUNK) {
    const int n0 = jt * NT;
    __syncthreads();   // Ss / B buffers / xsqt free from previous tile

    if (tid < NT) {
      const int ng = n0 + tid;
      xsqt[tid] = (ng < nfit) ? xsq[ng] : BIGF;
    }
    // stage B K-slice 0 into buffer 0
#if HAVE_TDM
    if (wid == 0) tdm_load_b_slice(xb, n0, nfit - n0, 0, bs_off);
#else
    {
      const int nl = tid >> 1, half = tid & 1;
      int ng = n0 + nl; if (ng >= nfit) ng = nfit - 1;
      const uint4* src = (const uint4*)(xb + ((size_t)ng << 9) + (half << 4));
      uint4* dst = (uint4*)(Bs + nl * BSTRIDE + (half << 4));
      dst[0] = src[0]; dst[1] = src[1];
    }
#endif

    v8f acc[2][4] = {};
    for (int ks = 0; ks < KSTEPS; ks++) {
#if HAVE_TDM
      if (wid == 0) __builtin_amdgcn_s_wait_tensorcnt(0);  // slice ks DMA complete
      __syncthreads();                                     // publish to all waves
      if (ks + 1 < KSTEPS && wid == 0)                     // DMA next slice into other buffer
        tdm_load_b_slice(xb, n0, nfit - n0, (ks + 1) * KT,
                         bs_off + ((ks + 1) & 1) * BBUF_BYTES);
#else
      __syncthreads();   // staged slice visible; prev compute done
      if (ks + 1 < KSTEPS) {   // prefetch next K-slice into the other buffer
        const int nl = tid >> 1, half = tid & 1;
        int ng = n0 + nl; if (ng >= nfit) ng = nfit - 1;
        const uint4* src = (const uint4*)(xb + ((size_t)ng << 9) + ((ks + 1) << 5) + (half << 4));
        uint4* dst = (uint4*)(Bs + ((ks + 1) & 1) * (NT * BSTRIDE) + nl * BSTRIDE + (half << 4));
        dst[0] = src[0]; dst[1] = src[1];
      }
#endif
      const unsigned short* Bc = Bs + (ks & 1) * (NT * BSTRIDE);
      const int k0 = ks * KT;

      v16bf af[2];
#pragma unroll
      for (int i = 0; i < 2; i++)
        af[i] = load_a_frag(As + (mw + i * 16 + lo) * ASTRIDE + k0 + hi * 8);
      v16bf bfm[4];
#pragma unroll
      for (int j = 0; j < 4; j++)
        bfm[j] = load_b_frag(Bc + (nw + j * 16 + lo) * BSTRIDE + hi * 8);
#pragma unroll
      for (int i = 0; i < 2; i++)
#pragma unroll
        for (int j = 0; j < 4; j++)
          acc[i][j] = __builtin_amdgcn_wmma_f32_16x16x32_bf16(
              false, af[i], false, bfm[j], (short)0, acc[i][j], false, false);
    }

    // Epilogue: score = ||x||^2 - 2*(q.x)  (q_sq dropped: constant per row, ranking-invariant)
#pragma unroll
    for (int i = 0; i < 2; i++)
#pragma unroll
      for (int j = 0; j < 4; j++)
#pragma unroll
        for (int v = 0; v < 8; v++) {
          const int row = mw + i * 16 + v + hi * 8;   // C layout: VGPR v -> M=v / v+8
          const int col = nw + j * 16 + lo;           // lane -> N
          Ss[row * SSTRIDE + col] = xsqt[col] - 2.0f * acc[i][j][v];
        }
    __syncthreads();

    if (tid < MT) {   // one thread per query row scans this 128-wide tile
      const float* srow = Ss + tid * SSTRIDE;
      for (int c = 0; c < NT; c++) {
        const int ng = n0 + c;
        if (ng < nfit) topk_insert(ts, ti, srow[c], ng);
      }
    }
  }

  if (tid < MT) {   // per-(row, chunk) partial top-5 to workspace
    const size_t base = ((size_t)(m0 + tid) * NCHUNK + chunk) * TOPK;
#pragma unroll
    for (int k = 0; k < TOPK; k++) { ps[base + k] = ts[k]; pidx[base + k] = ti[k]; }
  }
}

// ---- merge partial top-5 lists, gather original f32 rows, average ----
__global__ __launch_bounds__(256)
void knn_merge(const float* __restrict__ ps, const int* __restrict__ pidx,
               const float* __restrict__ xfit, float* __restrict__ out, int nfit) {
  __shared__ int sid[TOPK];
  const int b = blockIdx.x;
  if (threadIdx.x == 0) {
    float ts[TOPK] = {BIGF, BIGF, BIGF, BIGF, BIGF};
    int   ti[TOPK] = {0, 0, 0, 0, 0};
    const size_t base = (size_t)b * NCHUNK * TOPK;
    for (int c = 0; c < NCHUNK * TOPK; c++)
      topk_insert(ts, ti, ps[base + c], pidx[base + c]);
#pragma unroll
    for (int k = 0; k < TOPK; k++) {
      int id = ti[k];
      if (id < 0) id = 0;
      if (id >= nfit) id = nfit - 1;
      sid[k] = id;
    }
  }
  __syncthreads();
  for (int col = threadIdx.x; col < DDIM; col += blockDim.x) {
    float s = 0.f;
#pragma unroll
    for (int k = 0; k < TOPK; k++) s += xfit[((size_t)sid[k] << 9) + col];
    out[((size_t)b << 9) + col] = s * (1.0f / TOPK);
  }
}

extern "C" void kernel_launch(void* const* d_in, const int* in_sizes, int n_in,
                              void* d_out, int out_size, void* d_ws, size_t ws_size,
                              hipStream_t stream) {
  (void)n_in; (void)out_size; (void)ws_size;
  const float* q    = (const float*)d_in[0];   // x_enc [B,1,512] f32
  const float* xfit = (const float*)d_in[1];   // X_fit [N,512]  f32
  const int B = in_sizes[0] / DDIM;            // 2048
  const int N = in_sizes[1] / DDIM;            // 100000
  const int ntiles = (N + NT - 1) / NT;        // 782
  const int mtiles = B / MT;                   // 16

  // Workspace layout
  auto al = [](size_t x) { return (x + 255) & ~(size_t)255; };
  char* ws = (char*)d_ws;
  size_t off = 0;
  float*          xsq = (float*)(ws + off);          off += al((size_t)N * 4);
  unsigned short* xb  = (unsigned short*)(ws + off); off += al((size_t)N * DDIM * 2);
  float*          psc = (float*)(ws + off);          off += al((size_t)B * NCHUNK * TOPK * 4);
  int*            pid = (int*)(ws + off);

  hipFuncSetAttribute((const void*)knn_gemm_topk,
                      hipFuncAttributeMaxDynamicSharedMemorySize, SMEM_BYTES);

  knn_prep<<<(N + 7) / 8, 256, 0, stream>>>(xfit, xb, xsq, N);
  knn_gemm_topk<<<dim3(mtiles, NCHUNK), 256, SMEM_BYTES, stream>>>(
      q, xb, xsq, psc, pid, N, ntiles);
  knn_merge<<<B, 256, 0, stream>>>(psc, pid, xfit, (float*)d_out, N);
}
